// BiGAT_sGAT_1056561955274
// MI455X (gfx1250) — compile-verified, matrix operand order checked by reading, not run on hardware
//
#include <hip/hip_runtime.h>
#include <hip/hip_bf16.h>

// ---------------------------------------------------------------------------
// Vector types
// ---------------------------------------------------------------------------
typedef __attribute__((ext_vector_type(16))) __bf16        v16bf;
typedef __attribute__((ext_vector_type(8)))  float         v8f;
typedef __attribute__((ext_vector_type(4)))  float         f32x4;
typedef __attribute__((ext_vector_type(4)))  unsigned int  u32x4;
typedef __attribute__((ext_vector_type(2)))  unsigned int  u32x2;
typedef __attribute__((ext_vector_type(8)))  int           i32x8;
typedef __attribute__((ext_vector_type(4)))  int           i32x4;

union Frag16 {
    u32x4 u[2];   // two 16-byte LDS loads
    v16bf v;      // 16 bf16 elements for a WMMA A/B operand
};

__device__ __forceinline__ unsigned short f2bf(float f) {
    unsigned int u = __float_as_uint(f);
    u += 0x7fffu + ((u >> 16) & 1u);      // round-to-nearest-even
    return (unsigned short)(u >> 16);
}

// ---------------------------------------------------------------------------
// Tensor Data Mover: async 2D bf16 tile load  global -> LDS.
// Pads LDS rows: 16 dwords data + 4 dwords pad => 80B row stride (matches
// the sA[32][40] ushort layout and avoids LDS bank conflicts).
// ---------------------------------------------------------------------------
#if __has_builtin(__builtin_amdgcn_tensor_load_to_lds)
#define HAVE_TDM 1
__device__ __forceinline__ void
tdm_load_tile_bf16(unsigned long long ga, unsigned int ldsaddr,
                   unsigned int dim0, unsigned int dim1, unsigned int stride0) {
    u32x4 g0;
    g0[0] = 1u;                                       // count=1, user D#
    g0[1] = ldsaddr;                                  // lds_addr (bytes)
    g0[2] = (unsigned int)ga;                         // global_addr[31:0]
    g0[3] = (unsigned int)(ga >> 32) | (2u << 30);    // addr[56:32] | type=2
    i32x8 g1;
    g1[0] = (int)((1u << 16)      // data_size = 2 bytes
                | (1u << 20)      // pad_enable
                | (3u << 22)      // pad_interval: 16 dwords
                | (3u << 25));    // pad_amount:   4 dwords
    g1[1] = (int)((dim0 & 0xFFFFu) << 16);                       // tensor_dim0 lo
    g1[2] = (int)((dim0 >> 16) | ((dim1 & 0xFFFFu) << 16));      // dim0 hi | dim1 lo
    g1[3] = (int)((dim1 >> 16) | (32u << 16));                   // dim1 hi | tile_dim0=32
    g1[4] = (int)32u;                                            // tile_dim1=32
    g1[5] = (int)stride0;                                        // tensor_dim0_stride
    g1[6] = 0;
    g1[7] = 0;
    i32x4 gz = {0, 0, 0, 0};
#if defined(__clang_major__) && (__clang_major__ >= 23)
    i32x8 gz8 = {0, 0, 0, 0, 0, 0, 0, 0};
    __builtin_amdgcn_tensor_load_to_lds(g0, g1, gz, gz, gz8, 0);
#else
    __builtin_amdgcn_tensor_load_to_lds(g0, g1, gz, gz, 0);
#endif
}
#else
#define HAVE_TDM 0
#endif

// ---------------------------------------------------------------------------
// Weight prep: W[K][128] f32 -> Wt[128][K] bf16 (transpose + convert), once.
// ---------------------------------------------------------------------------
__global__ void transpose_f32_to_bf16(const float* __restrict__ W,
                                      unsigned short* __restrict__ Wt, int K) {
    int t = blockIdx.x * blockDim.x + threadIdx.x;
    if (t >= K * 128) return;
    int k = t >> 7, n = t & 127;
    Wt[(size_t)n * K + k] = f2bf(W[t]);
}

// ---------------------------------------------------------------------------
// GEMM: C[M,128] = A[M,K] @ W[K,128] (+bias).  W pre-transposed to bf16.
// Block = 256 threads (8 waves) computes a 32-row x 128-col tile; wave w owns
// columns [16w,16w+16) and two 16x16 row tiles sharing one B fragment:
// 2 x v_wmma_f32_16x16x32_bf16 per 32-wide K slab.
// A_BF16: A already bf16 [M][K] -> A tile staged by the Tensor Data Mover.
// C_BF16: emit bf16 output (feeds the next bf16-A GEMM).
// ---------------------------------------------------------------------------
template <bool A_BF16, bool C_BF16>
__global__ void __launch_bounds__(256)
gemm_wmma(const void* __restrict__ Av, const unsigned short* __restrict__ Wt,
          const float* __restrict__ bias, void* __restrict__ Cv, int M, int K) {
    __shared__ alignas(16) unsigned short sA[32][40];   // 80B row stride
    __shared__ alignas(16) unsigned short sB[128][40];

    const int tid  = threadIdx.x;
    const int wave = tid >> 5;
    const int lane = tid & 31;
    const int row0 = blockIdx.x << 5;
    const int m    = lane & 15;
    const int kh   = lane >> 4;                 // 0/1: K-half per lane group
    const int ncol = (wave << 4) + m;

    const float*          Af = (const float*)Av;
    const unsigned short* Ab = (const unsigned short*)Av;

    v8f acc0 = {}, acc1 = {};

    for (int k0 = 0; k0 < K; k0 += 32) {
        // ---- stage A tile (32 rows x 32 cols) into sA --------------------
        if constexpr (A_BF16) {
#if HAVE_TDM
            if (tid < 32) {
                unsigned long long ga = (unsigned long long)(uintptr_t)Ab +
                    ((unsigned long long)row0 * (unsigned)K + (unsigned)k0) * 2ull;
                tdm_load_tile_bf16(ga, (unsigned int)(uintptr_t)&sA[0][0],
                                   (unsigned int)(K - k0), (unsigned int)(M - row0),
                                   (unsigned int)K);
            }
#else
            {   // branchless copy: 256 x u32x2 (4 bf16 each)
                int r = tid >> 3, q = tid & 7;
                int gr  = row0 + r;
                int grc = (gr < M) ? gr : (M - 1);
                u32x2 v = *(const u32x2*)(Ab + (size_t)grc * K + k0 + q * 4);
                if (gr >= M) { v[0] = 0u; v[1] = 0u; }
                *(u32x2*)&sA[r][q * 4] = v;
            }
#endif
        } else {
            // branchless f32 load + bf16 convert: 256 x float4
            int r = tid >> 3, q = tid & 7;
            int gr   = row0 + r;
            int grc  = (gr < M) ? gr : (M - 1);
            int col  = k0 + q * 4;                    // K is a multiple of 4
            int colc = (col < K) ? col : (K - 4);
            f32x4 f  = *(const f32x4*)(Af + (size_t)grc * K + colc);
            bool ok  = (gr < M) & (col < K);
            float x0 = ok ? f[0] : 0.0f, x1 = ok ? f[1] : 0.0f;
            float x2 = ok ? f[2] : 0.0f, x3 = ok ? f[3] : 0.0f;
            u32x2 p;
            p[0] = (unsigned)f2bf(x0) | ((unsigned)f2bf(x1) << 16);
            p[1] = (unsigned)f2bf(x2) | ((unsigned)f2bf(x3) << 16);
            *(u32x2*)&sA[r][q * 4] = p;
        }

        // ---- stage B tile: sB[n][k] <- Wt[n][k0+k], 128 rows x 32 --------
#pragma unroll
        for (int i = 0; i < 4; ++i) {
            int idx = tid + i * 256;                  // 1024 u32x2 total
            int n = idx >> 3, q = idx & 7;
            int col  = k0 + q * 4;
            int colc = (col < K) ? col : (K - 4);
            u32x2 v  = *(const u32x2*)(Wt + (size_t)n * K + colc);
            if (col >= K) { v[0] = 0u; v[1] = 0u; }
            *(u32x2*)&sB[n][q * 4] = v;
        }

#if HAVE_TDM
        if constexpr (A_BF16) {
            if (tid < 32) __builtin_amdgcn_s_wait_tensorcnt(0);
        }
#endif
        __syncthreads();

        // ---- fragments (documented 16-bit 16x32 layout) ------------------
        // lanes 0-15: K=0..7 (v0-3), 16..23 (v4-7); lanes 16-31: +8
        Frag16 fa0, fa1, fb;
        fa0.u[0] = *(const u32x4*)&sA[m][kh * 8];
        fa0.u[1] = *(const u32x4*)&sA[m][16 + kh * 8];
        fa1.u[0] = *(const u32x4*)&sA[16 + m][kh * 8];
        fa1.u[1] = *(const u32x4*)&sA[16 + m][16 + kh * 8];
        fb.u[0]  = *(const u32x4*)&sB[ncol][kh * 8];
        fb.u[1]  = *(const u32x4*)&sB[ncol][16 + kh * 8];

        acc0 = __builtin_amdgcn_wmma_f32_16x16x32_bf16(
            false, fa0.v, false, fb.v, (short)0, acc0, false, false);
        acc1 = __builtin_amdgcn_wmma_f32_16x16x32_bf16(
            false, fa1.v, false, fb.v, (short)0, acc1, false, false);
        __syncthreads();
    }

    // ---- epilogue: C/D layout lane<16 -> M=v, lane>=16 -> M=8+v ----------
    const float bv = bias ? bias[ncol] : 0.0f;
#pragma unroll
    for (int v = 0; v < 8; ++v) {
        int r0 = row0 + v + (kh << 3);
        int r1 = r0 + 16;
        if constexpr (C_BF16) {
            unsigned short* C = (unsigned short*)Cv;
            if (r0 < M) C[(size_t)r0 * 128 + ncol] = f2bf(acc0[v] + bv);
            if (r1 < M) C[(size_t)r1 * 128 + ncol] = f2bf(acc1[v] + bv);
        } else {
            float* C = (float*)Cv;
            if (r0 < M) C[(size_t)r0 * 128 + ncol] = acc0[v] + bv;
            if (r1 < M) C[(size_t)r1 * 128 + ncol] = acc1[v] + bv;
        }
    }
}

// ---------------------------------------------------------------------------
// Init accumulators (d_ws is poisoned by the harness)
// ---------------------------------------------------------------------------
__global__ void init_ws(float* mx, float* denom, float* acc, int N) {
    int t = blockIdx.x * blockDim.x + threadIdx.x;
    if (t < N * 128) acc[t] = 0.0f;
    if (t < N * 4) { mx[t] = __int_as_float(0xff800000); denom[t] = 0.0f; }
}

// ---------------------------------------------------------------------------
// Pass 1: per-edge logits e[e][h] = att[h,:] . LeakyReLU(xl[src]+xr[dst])
// One wave32 per edge; lane = channel; shfl reduction; f32 atomic max.
// ---------------------------------------------------------------------------
__global__ void __launch_bounds__(256)
edge_logits(const float* __restrict__ xl, const float* __restrict__ xr,
            const float* __restrict__ att, const int* __restrict__ ei,
            float* __restrict__ ebuf, float* __restrict__ mx, int E) {
    int e = (blockIdx.x * 256 + threadIdx.x) >> 5;
    int lane = threadIdx.x & 31;
    if (e >= E) return;
    int src = ei[e], dst = ei[E + e];
#pragma unroll
    for (int h = 0; h < 4; ++h) {
        int c = (h << 5) + lane;
        float v = xl[(size_t)src * 128 + c] + xr[(size_t)dst * 128 + c];
        v = (v > 0.0f) ? v : 0.2f * v;           // LeakyReLU(0.2)
        float p = att[c] * v;
#pragma unroll
        for (int off = 16; off; off >>= 1) p += __shfl_xor(p, off, 32);
        if (lane == 0) {
            ebuf[e * 4 + h] = p;
            atomicMax(&mx[dst * 4 + h], p);
        }
    }
}

// ---------------------------------------------------------------------------
// Pass 2: ex = exp(e - m[dst]); accumulate per-dst denominators.
// ---------------------------------------------------------------------------
__global__ void edge_softmax_num(const int* __restrict__ ei, float* __restrict__ ebuf,
                                 const float* __restrict__ mx, float* __restrict__ denom,
                                 int E) {
    int t = blockIdx.x * blockDim.x + threadIdx.x;
    if (t >= E * 4) return;
    int e = t >> 2, h = t & 3;
    int dst = ei[E + e];
    float ex = __expf(ebuf[t] - mx[dst * 4 + h]);
    ebuf[t] = ex;
    atomicAdd(&denom[dst * 4 + h], ex);
}

// ---------------------------------------------------------------------------
// Pass 3: acc[dst] += alpha * xl[src]; one wave per edge, lane = channel.
// ---------------------------------------------------------------------------
__global__ void __launch_bounds__(256)
edge_scatter(const float* __restrict__ xl, const int* __restrict__ ei,
             const float* __restrict__ ebuf, const float* __restrict__ denom,
             float* __restrict__ acc, int E) {
    int e = (blockIdx.x * 256 + threadIdx.x) >> 5;
    int lane = threadIdx.x & 31;
    if (e >= E) return;
    int src = ei[e], dst = ei[E + e];
#pragma unroll
    for (int h = 0; h < 4; ++h) {
        float alpha = ebuf[e * 4 + h] / (denom[dst * 4 + h] + 1e-16f);
        int c = (h << 5) + lane;
        atomicAdd(&acc[(size_t)dst * 128 + c], alpha * xl[(size_t)src * 128 + c]);
    }
}

// ---------------------------------------------------------------------------
// Final: out = ELU(acc + b_gat)
// ---------------------------------------------------------------------------
__global__ void finalize(const float* __restrict__ acc, const float* __restrict__ b_gat,
                         float* __restrict__ out, int total) {
    int t = blockIdx.x * blockDim.x + threadIdx.x;
    if (t >= total) return;
    float v = acc[t] + b_gat[t & 127];
    out[t] = (v > 0.0f) ? v : (__expf(v) - 1.0f);
}

// ---------------------------------------------------------------------------
// Host launcher
// ---------------------------------------------------------------------------
extern "C" void kernel_launch(void* const* d_in, const int* in_sizes, int n_in,
                              void* d_out, int out_size, void* d_ws, size_t ws_size,
                              hipStream_t stream) {
    const float* x_mrna  = (const float*)d_in[0];
    const float* x_mirna = (const float*)d_in[1];
    const float* Wp1     = (const float*)d_in[2];
    const float* bp1     = (const float*)d_in[3];
    const float* Wp2     = (const float*)d_in[4];
    const float* bp2     = (const float*)d_in[5];
    const float* Wl      = (const float*)d_in[6];
    const float* Wr      = (const float*)d_in[7];
    const float* att     = (const float*)d_in[8];
    const float* b_gat   = (const float*)d_in[9];
    const int*   ei      = (const int*)d_in[10];

    constexpr int N1 = 25000, N2 = 25000, N = N1 + N2, E = 800000;
    constexpr int D1 = 2000, D2 = 500;

    char* ws = (char*)d_ws;
    size_t off = 0;
    auto alloc = [&](size_t bytes) { char* p = ws + off; off += (bytes + 255) & ~(size_t)255; return p; };

    float*          xl     = (float*)alloc((size_t)N * 128 * 4);
    float*          xr     = (float*)alloc((size_t)N * 128 * 4);
    float*          ebuf   = (float*)alloc((size_t)E * 4 * 4);
    float*          mx     = (float*)alloc((size_t)N * 4 * 4);
    float*          denom  = (float*)alloc((size_t)N * 4 * 4);
    float*          acc    = (float*)alloc((size_t)N * 128 * 4);
    unsigned short* xprojb = (unsigned short*)alloc((size_t)N * 128 * 2);
    unsigned short* Wt1    = (unsigned short*)alloc((size_t)128 * D1 * 2);
    unsigned short* Wt2    = (unsigned short*)alloc((size_t)128 * D2 * 2);
    unsigned short* Wtl    = (unsigned short*)alloc((size_t)128 * 128 * 2);
    unsigned short* Wtr    = (unsigned short*)alloc((size_t)128 * 128 * 2);

    dim3 blk(256);

    // Weight prep: transpose + bf16 convert (tiny, one-shot per call)
    transpose_f32_to_bf16<<<dim3((D1 * 128 + 255) / 256), blk, 0, stream>>>(Wp1, Wt1, D1);
    transpose_f32_to_bf16<<<dim3((D2 * 128 + 255) / 256), blk, 0, stream>>>(Wp2, Wt2, D2);
    transpose_f32_to_bf16<<<dim3((128 * 128 + 255) / 256), blk, 0, stream>>>(Wl, Wtl, 128);
    transpose_f32_to_bf16<<<dim3((128 * 128 + 255) / 256), blk, 0, stream>>>(Wr, Wtr, 128);

    // Projections: f32 A -> bf16 xproj (memory-bound; WMMA bf16, f32 acc)
    gemm_wmma<false, true><<<dim3((N1 + 31) / 32), blk, 0, stream>>>(x_mrna,  Wt1, bp1, xprojb, N1, D1);
    gemm_wmma<false, true><<<dim3((N2 + 31) / 32), blk, 0, stream>>>(x_mirna, Wt2, bp2, xprojb + (size_t)N1 * 128, N2, D2);
    // GATv2 transforms: bf16 A (TDM-staged) -> f32 xl/xr
    gemm_wmma<true, false><<<dim3((N + 31) / 32), blk, 0, stream>>>(xprojb, Wtl, nullptr, xl, N, 128);
    gemm_wmma<true, false><<<dim3((N + 31) / 32), blk, 0, stream>>>(xprojb, Wtr, nullptr, xr, N, 128);

    // Edge phase (gather/atomic bound — the real bottleneck)
    init_ws<<<dim3((N * 128 + 255) / 256), blk, 0, stream>>>(mx, denom, acc, N);
    edge_logits<<<dim3((E + 7) / 8), blk, 0, stream>>>(xl, xr, att, ei, ebuf, mx, E);
    edge_softmax_num<<<dim3((E * 4 + 255) / 256), blk, 0, stream>>>(ei, ebuf, mx, denom, E);
    edge_scatter<<<dim3((E + 7) / 8), blk, 0, stream>>>(xl, ei, ebuf, denom, acc, E);
    finalize<<<dim3((N * 128 + 255) / 256), blk, 0, stream>>>(acc, b_gat, (float*)d_out, N * 128);
}
